// EncoderBlock_29008209118036
// MI455X (gfx1250) — compile-verified, matrix operand order checked by reading, not run on hardware
//
#include <hip/hip_runtime.h>
#include <math.h>

typedef __attribute__((ext_vector_type(16))) __bf16 v16bf;
typedef __attribute__((ext_vector_type(8)))  __bf16 v8bf;
typedef __attribute__((ext_vector_type(8)))  float  v8f;
typedef __attribute__((ext_vector_type(4)))  unsigned int u32x4;
typedef __attribute__((ext_vector_type(8)))  int i32x8;
typedef __attribute__((ext_vector_type(4)))  int i32x4;

#define D_MODEL 1024
#define FFN     4096
#define NHEAD   16
#define DKH     64

// ---------------------------------------------------------------------------
// TDM: DMA a [128 rows x 32 bf16] tile (row stride = rowlen elems in memory)
// into LDS at lds_off, inserting 16B of LDS padding after every 64B row so the
// LDS row stride is 40 halves (bank-conflict-free ds_read_b128 fragments).
// D# fields per CDNA5 ISA #8.3/#8.4.
// ---------------------------------------------------------------------------
__device__ __forceinline__ void tdm_load_tile_128x32(
    unsigned int lds_off, const __bf16* gbase, unsigned long long elem_off,
    int rowlen, int nrows_total) {
  unsigned long long ga = (unsigned long long)(uintptr_t)gbase + elem_off * 2ull;
  u32x4 g0;
  g0[0] = 1u;                                   // count=1, user mode
  g0[1] = lds_off;                              // lds_addr
  g0[2] = (unsigned int)(ga & 0xffffffffull);   // global_addr[31:0]
  g0[3] = (unsigned int)((ga >> 32) & 0x1ffffffull) | (2u << 30);  // type=2
  i32x8 g1;
  // data_size=1 (2B), pad_enable=1, pad_interval=3 (16 DW = 64B), pad_amount=3 (4 DW = 16B)
  g1[0] = (1 << 16) | (1 << 20) | (3 << 22) | (3 << 25);
  g1[1] = (rowlen & 0xffff) << 16;                                   // tensor_dim0[15:0]
  g1[2] = (int)(((unsigned)rowlen >> 16) |
                (((unsigned)nrows_total & 0xffff) << 16));           // dim0 hi | dim1 lo
  g1[3] = (int)(((unsigned)nrows_total >> 16) | (32u << 16));        // dim1 hi | tile_dim0=32
  g1[4] = 128;                                                       // tile_dim1=128
  g1[5] = rowlen;                                                    // dim0_stride[31:0]
  g1[6] = 0;
  g1[7] = 0;
  i32x4 z4 = {0, 0, 0, 0};
#if __clang_major__ >= 23
  i32x8 z8 = {0, 0, 0, 0, 0, 0, 0, 0};
  __builtin_amdgcn_tensor_load_to_lds(g0, g1, z4, z4, z8, 0);
#else
  __builtin_amdgcn_tensor_load_to_lds(g0, g1, z4, z4, 0);
#endif
}

// ---------------------------------------------------------------------------
// Weight transpose+convert: src f32 [K][N] -> dst bf16 [N][K]
// ---------------------------------------------------------------------------
__global__ __launch_bounds__(256) void cvt_transpose_bf16(
    const float* __restrict__ src, __bf16* __restrict__ dst, int K, int N) {
  __shared__ float tile[32][33];
  const int kb = blockIdx.x * 32, nb = blockIdx.y * 32;
  const int tx = threadIdx.x & 31, ty = threadIdx.x >> 5;  // ty: 0..7
#pragma unroll
  for (int i = 0; i < 4; ++i)
    tile[ty + i * 8][tx] = src[(size_t)(kb + ty + i * 8) * N + nb + tx];
  __syncthreads();
#pragma unroll
  for (int i = 0; i < 4; ++i)
    dst[(size_t)(nb + ty + i * 8) * K + kb + tx] = (__bf16)tile[tx][ty + i * 8];
}

// ---------------------------------------------------------------------------
// LayerNorm over D_MODEL, one block per row, bf16 output
// ---------------------------------------------------------------------------
__global__ __launch_bounds__(256) void layernorm_bf16(const float* __restrict__ X,
                                                      const float* __restrict__ g,
                                                      const float* __restrict__ b,
                                                      __bf16* __restrict__ Y) {
  const int D = D_MODEL;
  const int row = blockIdx.x;
  const float* x = X + (size_t)row * D;
  float s = 0.f, ss = 0.f;
  for (int i = threadIdx.x; i < D; i += 256) { float v = x[i]; s += v; ss += v * v; }
#pragma unroll
  for (int m = 16; m >= 1; m >>= 1) { s += __shfl_xor(s, m); ss += __shfl_xor(ss, m); }
  __shared__ float red[2][8];
  const int wave = threadIdx.x >> 5;
  if ((threadIdx.x & 31) == 0) { red[0][wave] = s; red[1][wave] = ss; }
  __syncthreads();
  s = 0.f; ss = 0.f;
#pragma unroll
  for (int w = 0; w < 8; ++w) { s += red[0][w]; ss += red[1][w]; }
  const float mu  = s / D;
  const float inv = rsqrtf(ss / D - mu * mu + 1e-6f);
  __bf16* y = Y + (size_t)row * D;
  for (int i = threadIdx.x; i < D; i += 256)
    y[i] = (__bf16)((x[i] - mu) * inv * g[i] + b[i]);
}

// ---------------------------------------------------------------------------
// Tiled WMMA GEMM: C[M,N] = A[M,K](bf16, row-major) x Bt[N,K](bf16, K-major)
// Block = 256 threads (8 waves), macro tile 128x128, wave tile 32x64, TK=32.
// Both tiles DMA'd by the Tensor Data Mover into double-buffered padded LDS
// (issued by wave 0; TENSORcnt in-order per wave -> wait<=2 overlaps next DMA
// with compute). Fragments: two ds_read_b128 per 16x32 fragment.
// ---------------------------------------------------------------------------
template <bool BIAS, bool RELU, bool RES, bool OUTBF>
__global__ __launch_bounds__(256) void gemm_wmma_bf16(
    const __bf16* __restrict__ A, const __bf16* __restrict__ Bt,
    const float* __restrict__ bias, const float* __restrict__ res,
    float* __restrict__ Cf, __bf16* __restrict__ Cb, int M, int N, int K) {
  constexpr int TM = 128, TN = 128, TK = 32, LSTR = 40;
  __shared__ __align__(16) __bf16 As[2][TM * LSTR];
  __shared__ __align__(16) __bf16 Bs[2][TN * LSTR];
  const int tid = threadIdx.x;
  const int wave = tid >> 5, lane = tid & 31;
  const int bm = blockIdx.x * TM, bn = blockIdx.y * TN;
  const int wm = (wave & 3) * 32, wn = (wave >> 2) * 64;
  const int l15 = lane & 15, kh = (lane >> 4) * 8;
  const int nk = K / TK;

  const unsigned int ldsA[2] = {(unsigned int)(uintptr_t)&As[0][0],
                                (unsigned int)(uintptr_t)&As[1][0]};
  const unsigned int ldsB[2] = {(unsigned int)(uintptr_t)&Bs[0][0],
                                (unsigned int)(uintptr_t)&Bs[1][0]};

  if (wave == 0) {
    tdm_load_tile_128x32(ldsA[0], A,  (unsigned long long)bm * K, K, M);
    tdm_load_tile_128x32(ldsB[0], Bt, (unsigned long long)bn * K, K, N);
  }

  v8f acc[2][4] = {};
  for (int t = 0; t < nk; ++t) {
    const int buf = t & 1;
    if (wave == 0) {
      if (t + 1 < nk) {
        tdm_load_tile_128x32(ldsA[buf ^ 1], A,
                             (unsigned long long)bm * K + (t + 1) * TK, K, M);
        tdm_load_tile_128x32(ldsB[buf ^ 1], Bt,
                             (unsigned long long)bn * K + (t + 1) * TK, K, N);
        __builtin_amdgcn_s_wait_tensorcnt(2);  // tile t landed; t+1 in flight
      } else {
        __builtin_amdgcn_s_wait_tensorcnt(0);
      }
    }
    __syncthreads();

    v16bf af[2], bf[4];
#pragma unroll
    for (int i = 0; i < 2; ++i) {
      const __bf16* p = &As[buf][0] + (wm + i * 16 + l15) * LSTR;
      ((v8bf*)&af[i])[0] = *(const v8bf*)(p + kh);
      ((v8bf*)&af[i])[1] = *(const v8bf*)(p + 16 + kh);
    }
#pragma unroll
    for (int j = 0; j < 4; ++j) {
      const __bf16* p = &Bs[buf][0] + (wn + j * 16 + l15) * LSTR;
      ((v8bf*)&bf[j])[0] = *(const v8bf*)(p + kh);
      ((v8bf*)&bf[j])[1] = *(const v8bf*)(p + 16 + kh);
    }
#pragma unroll
    for (int i = 0; i < 2; ++i)
#pragma unroll
      for (int j = 0; j < 4; ++j)
        acc[i][j] = __builtin_amdgcn_wmma_f32_16x16x32_bf16(
            false, af[i], false, bf[j], (short)0, acc[i][j], false, false);
    __syncthreads();  // all waves done reading buf before it is overwritten
  }

  const int mh = (lane >> 4) * 8;
#pragma unroll
  for (int j = 0; j < 4; ++j) {
    const int n = bn + wn + j * 16 + l15;
    const float bv = BIAS ? bias[n] : 0.f;
#pragma unroll
    for (int i = 0; i < 2; ++i) {
#pragma unroll
      for (int e = 0; e < 8; ++e) {
        const int m = bm + wm + i * 16 + mh + e;
        float v = acc[i][j][e] + bv;
        if (RELU) v = fmaxf(v, 0.f);
        if (RES) v += res[(size_t)m * N + n];
        if (OUTBF) Cb[(size_t)m * N + n] = (__bf16)v;
        else       Cf[(size_t)m * N + n] = v;
      }
    }
  }
}

// ---------------------------------------------------------------------------
// Flash attention (bf16 QKV, bf16 ctx out). Block = (b,h, 128-query chunk),
// 8 waves x 16 queries. Online softmax; 32-key tiles staged in LDS.
// ---------------------------------------------------------------------------
__global__ __launch_bounds__(256) void flash_attn_bf16(
    const __bf16* __restrict__ Q, const __bf16* __restrict__ Km,
    const __bf16* __restrict__ V, __bf16* __restrict__ O, int S) {
  constexpr int KSTR = 72, VSTR = 40, PSTR = 40;
  __shared__ __align__(16) __bf16 Ks[32 * KSTR];    // [key][d]
  __shared__ __align__(16) __bf16 Vt[64 * VSTR];    // [d][key]
  __shared__ __align__(16) __bf16 Ps[8][16 * PSTR]; // per-wave P scratch [m][key]
  const int tid = threadIdx.x, wave = tid >> 5, lane = tid & 31;
  const int l15 = lane & 15, kh = (lane >> 4) * 8, mh = (lane >> 4) * 8;
  const int h = blockIdx.x & (NHEAD - 1);
  const int b = blockIdx.x >> 4;
  const size_t rowbase = (size_t)b * S;
  const int q0 = blockIdx.y * 128 + wave * 16;

  v16bf qf[2];
  const __bf16* qp = Q + (rowbase + q0 + l15) * D_MODEL + h * DKH;
#pragma unroll
  for (int d = 0; d < 2; ++d) {
    ((v8bf*)&qf[d])[0] = *(const v8bf*)(qp + d * 32 + kh);
    ((v8bf*)&qf[d])[1] = *(const v8bf*)(qp + d * 32 + 16 + kh);
  }
  v8f oacc[4] = {};
  v8f mrow, lrow;
#pragma unroll
  for (int e = 0; e < 8; ++e) { mrow[e] = -1e30f; lrow[e] = 0.f; }

  for (int kt = 0; kt < S; kt += 32) {
    __syncthreads();
    {  // cooperative K/V tile load: 256 threads x one 16B chunk each
      const int key = tid >> 3, dc = (tid & 7) * 8;
      *(v8bf*)(Ks + key * KSTR + dc) =
          *(const v8bf*)(Km + (rowbase + kt + key) * D_MODEL + h * DKH + dc);
      v8bf dv = *(const v8bf*)(V + (rowbase + kt + key) * D_MODEL + h * DKH + dc);
#pragma unroll
      for (int e = 0; e < 8; ++e) Vt[(dc + e) * VSTR + key] = dv[e];
    }
    __syncthreads();
    // scores S16x32 = Q(16x64) * K^T, 2 key subtiles, depth 64
    v8f s[2];
#pragma unroll
    for (int j = 0; j < 2; ++j) {
      const __bf16* kp = Ks + (j * 16 + l15) * KSTR;
      v16bf kf0, kf1;
      ((v8bf*)&kf0)[0] = *(const v8bf*)(kp + kh);
      ((v8bf*)&kf0)[1] = *(const v8bf*)(kp + 16 + kh);
      ((v8bf*)&kf1)[0] = *(const v8bf*)(kp + 32 + kh);
      ((v8bf*)&kf1)[1] = *(const v8bf*)(kp + 48 + kh);
      v8f t = {};
      t = __builtin_amdgcn_wmma_f32_16x16x32_bf16(false, qf[0], false, kf0,
                                                  (short)0, t, false, false);
      t = __builtin_amdgcn_wmma_f32_16x16x32_bf16(false, qf[1], false, kf1,
                                                  (short)0, t, false, false);
      s[j] = t * 0.125f;  // 1/sqrt(64)
    }
    // online softmax: row reductions across 16 lanes of each half-wave
    v8f tmax;
#pragma unroll
    for (int e = 0; e < 8; ++e) tmax[e] = fmaxf(s[0][e], s[1][e]);
#pragma unroll
    for (int m = 1; m <= 8; m <<= 1)
#pragma unroll
      for (int e = 0; e < 8; ++e) tmax[e] = fmaxf(tmax[e], __shfl_xor(tmax[e], m));
    v8f mnew, corr, rsum;
#pragma unroll
    for (int e = 0; e < 8; ++e) {
      mnew[e] = fmaxf(mrow[e], tmax[e]);
      corr[e] = __expf(mrow[e] - mnew[e]);
      rsum[e] = 0.f;
    }
#pragma unroll
    for (int j = 0; j < 2; ++j)
#pragma unroll
      for (int e = 0; e < 8; ++e) {
        float p = __expf(s[j][e] - mnew[e]);
        s[j][e] = p;
        rsum[e] += p;
      }
#pragma unroll
    for (int m = 1; m <= 8; m <<= 1)
#pragma unroll
      for (int e = 0; e < 8; ++e) rsum[e] += __shfl_xor(rsum[e], m);
#pragma unroll
    for (int e = 0; e < 8; ++e) { lrow[e] = lrow[e] * corr[e] + rsum[e]; mrow[e] = mnew[e]; }
#pragma unroll
    for (int d = 0; d < 4; ++d)
#pragma unroll
      for (int e = 0; e < 8; ++e) oacc[d][e] *= corr[e];
    // reshape P (C-layout -> A-fragment) via per-wave LDS scratch
    __bf16* ps = Ps[wave];
#pragma unroll
    for (int j = 0; j < 2; ++j)
#pragma unroll
      for (int e = 0; e < 8; ++e)
        ps[(mh + e) * PSTR + j * 16 + l15] = (__bf16)s[j][e];
    v16bf pf;
    const __bf16* pp = ps + l15 * PSTR;
    ((v8bf*)&pf)[0] = *(const v8bf*)(pp + kh);
    ((v8bf*)&pf)[1] = *(const v8bf*)(pp + 16 + kh);
    // ctx += P(16x32) * V(32x64)
#pragma unroll
    for (int d = 0; d < 4; ++d) {
      const __bf16* vp = Vt + (d * 16 + l15) * VSTR;
      v16bf vf;
      ((v8bf*)&vf)[0] = *(const v8bf*)(vp + kh);
      ((v8bf*)&vf)[1] = *(const v8bf*)(vp + 16 + kh);
      oacc[d] = __builtin_amdgcn_wmma_f32_16x16x32_bf16(false, pf, false, vf,
                                                        (short)0, oacc[d], false, false);
    }
  }
#pragma unroll
  for (int d = 0; d < 4; ++d)
#pragma unroll
    for (int e = 0; e < 8; ++e) {
      float val = oacc[d][e] / lrow[e];
      O[(rowbase + q0 + mh + e) * D_MODEL + h * DKH + d * 16 + l15] = (__bf16)val;
    }
}

// ---------------------------------------------------------------------------
// host-side orchestration
// ---------------------------------------------------------------------------
extern "C" void kernel_launch(void* const* d_in, const int* in_sizes, int n_in,
                              void* d_out, int out_size, void* d_ws, size_t ws_size,
                              hipStream_t stream) {
  const float* x    = (const float*)d_in[0];
  const float* Wq   = (const float*)d_in[1];
  const float* Wk   = (const float*)d_in[2];
  const float* Wv   = (const float*)d_in[3];
  const float* Wo   = (const float*)d_in[4];
  const float* b_o  = (const float*)d_in[5];
  const float* W1   = (const float*)d_in[6];
  const float* b1   = (const float*)d_in[7];
  const float* W2   = (const float*)d_in[8];
  const float* b2   = (const float*)d_in[9];
  const float* ln1g = (const float*)d_in[10];
  const float* ln1b = (const float*)d_in[11];
  const float* ln2g = (const float*)d_in[12];
  const float* ln2b = (const float*)d_in[13];

  const int M = in_sizes[0] / D_MODEL;  // B*S = 4096
  const int S = 2048;
  const int B = M / S;

  char* w = (char*)d_ws;
  auto carve = [&](size_t bytes) -> char* {
    char* p = w;
    w += (bytes + 255) & ~(size_t)255;
    return p;
  };
  // transposed bf16 weights: [N][K]
  __bf16* Wq_t  = (__bf16*)carve((size_t)D_MODEL * D_MODEL * 2);
  __bf16* Wk_t  = (__bf16*)carve((size_t)D_MODEL * D_MODEL * 2);
  __bf16* Wv_t  = (__bf16*)carve((size_t)D_MODEL * D_MODEL * 2);
  __bf16* Wo_t  = (__bf16*)carve((size_t)D_MODEL * D_MODEL * 2);
  __bf16* W1_t  = (__bf16*)carve((size_t)D_MODEL * FFN * 2);
  __bf16* W2_t  = (__bf16*)carve((size_t)FFN * D_MODEL * 2);
  __bf16* h_b   = (__bf16*)carve((size_t)M * D_MODEL * 2);
  __bf16* q_b   = (__bf16*)carve((size_t)M * D_MODEL * 2);
  __bf16* k_b   = (__bf16*)carve((size_t)M * D_MODEL * 2);
  __bf16* v_b   = (__bf16*)carve((size_t)M * D_MODEL * 2);
  __bf16* ctx_b = (__bf16*)carve((size_t)M * D_MODEL * 2);
  float*  x1    = (float*) carve((size_t)M * D_MODEL * 4);
  __bf16* h2_b  = (__bf16*)carve((size_t)M * D_MODEL * 2);
  __bf16* f1_b  = (__bf16*)carve((size_t)M * FFN * 2);

  // weight transpose + bf16 convert (src [K][N] -> dst [N][K])
  cvt_transpose_bf16<<<dim3(D_MODEL / 32, D_MODEL / 32), 256, 0, stream>>>(Wq, Wq_t, D_MODEL, D_MODEL);
  cvt_transpose_bf16<<<dim3(D_MODEL / 32, D_MODEL / 32), 256, 0, stream>>>(Wk, Wk_t, D_MODEL, D_MODEL);
  cvt_transpose_bf16<<<dim3(D_MODEL / 32, D_MODEL / 32), 256, 0, stream>>>(Wv, Wv_t, D_MODEL, D_MODEL);
  cvt_transpose_bf16<<<dim3(D_MODEL / 32, D_MODEL / 32), 256, 0, stream>>>(Wo, Wo_t, D_MODEL, D_MODEL);
  cvt_transpose_bf16<<<dim3(D_MODEL / 32, FFN / 32),     256, 0, stream>>>(W1, W1_t, D_MODEL, FFN);
  cvt_transpose_bf16<<<dim3(FFN / 32, D_MODEL / 32),     256, 0, stream>>>(W2, W2_t, FFN, D_MODEL);

  // LN1
  layernorm_bf16<<<M, 256, 0, stream>>>(x, ln1g, ln1b, h_b);

  // Q, K, V projections
  dim3 g1(M / 128, D_MODEL / 128);
  gemm_wmma_bf16<false, false, false, true><<<g1, 256, 0, stream>>>(
      h_b, Wq_t, nullptr, nullptr, nullptr, q_b, M, D_MODEL, D_MODEL);
  gemm_wmma_bf16<false, false, false, true><<<g1, 256, 0, stream>>>(
      h_b, Wk_t, nullptr, nullptr, nullptr, k_b, M, D_MODEL, D_MODEL);
  gemm_wmma_bf16<false, false, false, true><<<g1, 256, 0, stream>>>(
      h_b, Wv_t, nullptr, nullptr, nullptr, v_b, M, D_MODEL, D_MODEL);

  // attention
  dim3 ga(B * NHEAD, S / 128);
  flash_attn_bf16<<<ga, 256, 0, stream>>>(q_b, k_b, v_b, ctx_b, S);

  // output projection + bias + residual -> x1 (f32)
  gemm_wmma_bf16<true, false, true, false><<<g1, 256, 0, stream>>>(
      ctx_b, Wo_t, b_o, x, x1, nullptr, M, D_MODEL, D_MODEL);

  // LN2
  layernorm_bf16<<<M, 256, 0, stream>>>(x1, ln2g, ln2b, h2_b);

  // FFN1: relu(h2 @ W1 + b1) -> bf16
  dim3 g2(M / 128, FFN / 128);
  gemm_wmma_bf16<true, true, false, true><<<g2, 256, 0, stream>>>(
      h2_b, W1_t, b1, nullptr, nullptr, f1_b, M, FFN, D_MODEL);

  // FFN2: f1 @ W2 + b2 + x1 -> d_out (f32)
  gemm_wmma_bf16<true, false, true, false><<<g1, 256, 0, stream>>>(
      f1_b, W2_t, b2, x1, (float*)d_out, nullptr, M, D_MODEL, FFN);
}
